// PagedLLaMABlock_62414464745817
// MI455X (gfx1250) — matrix operand order repaired and checked by reading, not compiled, not run on hardware
//
#include <hip/hip_runtime.h>
#include <math.h>
#include <stdint.h>

// ---------------- problem dims ----------------
#define S_   2048
#define E_   4096
#define NH_  32
#define KVH_ 8
#define HD_  128
#define HID_ 11008
#define QKV_N (NH_*HD_ + 2*KVH_*HD_)   // 6144
#define EPSV 1e-5f

// ---------------- WMMA types ------------------
typedef __attribute__((ext_vector_type(16))) __bf16 v16bf;
typedef __attribute__((ext_vector_type(8)))  __bf16 v8bf;
typedef __attribute__((ext_vector_type(8)))  float  v8f;

union Frag { v16bf v; v8bf h[2]; };

// A-matrix 16x32 bf16 fragment from row-major [M][K] memory.
// Lanes 0-15 : row = lane,    v0-3 = K[k0..k0+7],   v4-7 = K[k0+16..k0+23]
// Lanes 16-31: row = lane-16, v0-3 = K[k0+8..k0+15],v4-7 = K[k0+24..k0+31]
__device__ inline Frag load_a_frag(const __bf16* base, int ld, int row0, int k0, int lane) {
    Frag f;
    int r = lane & 15, half = lane >> 4;
    const __bf16* p = base + (size_t)(row0 + r) * (size_t)ld + k0 + half * 8;
    f.h[0] = *(const v8bf*)p;
    f.h[1] = *(const v8bf*)(p + 16);
    return f;
}

// B-matrix 32x16 bf16 fragment from row-major [N][K] memory (i.e. B^T).
__device__ inline Frag load_b_frag(const __bf16* base, int ld, int n0, int k0, int lane) {
    Frag f;
    int r = lane & 15, half = lane >> 4;
    const __bf16* p = base + (size_t)(n0 + r) * (size_t)ld + k0 + half * 8;
    f.h[0] = *(const v8bf*)p;
    f.h[1] = *(const v8bf*)(p + 16);
    return f;
}

__device__ inline Frag load_b_frag_cl(const __bf16* base, int ld, int n0, int k0, int lane, int nmax) {
    Frag f;
    int r = lane & 15, half = lane >> 4;
    int n = n0 + r; if (n > nmax) n = nmax;
    const __bf16* p = base + (size_t)n * (size_t)ld + k0 + half * 8;
    f.h[0] = *(const v8bf*)p;
    f.h[1] = *(const v8bf*)(p + 16);
    return f;
}

__device__ inline v8f wmma_bf16(const Frag& a, const Frag& b, v8f c) {
    return __builtin_amdgcn_wmma_f32_16x16x32_bf16(false, a.v, false, b.v, (short)0, c, false, false);
}

// gfx1250 async Global->LDS copy of one 16-byte chunk (tracked by ASYNCcnt).
// Generic LDS pointers carry the allocation-relative offset in bits [31:0]
// (SHARED_BASE lives in [63:32]), so a 32-bit truncation is the LDS address.
__device__ inline void async_copy_b128(const __bf16* gp, __bf16* lp) {
    uint32_t lo = (uint32_t)(uintptr_t)lp;
    asm volatile("global_load_async_to_lds_b128 %0, %1, off"
                 :: "v"(lo), "v"(gp) : "memory");
}
__device__ inline void wait_async0() {
    asm volatile("s_wait_asynccnt 0x0" ::: "memory");
}

// ------------- weight transpose + fp32->bf16 -------------
// dst[n*K + k] = (bf16) src[k*N + n].  K, N multiples of 32.
__global__ __launch_bounds__(256) void transpose_cvt(const float* __restrict__ src,
                                                     __bf16* __restrict__ dst,
                                                     int K, int N) {
    __shared__ float tile[32][33];
    int tx = threadIdx.x, ty = threadIdx.y;          // block (32,8)
    int n0 = blockIdx.x * 32, k0 = blockIdx.y * 32;
    for (int i = 0; i < 4; ++i) {
        int k = k0 + ty + i * 8;
        tile[ty + i * 8][tx] = src[(size_t)k * N + (n0 + tx)];
    }
    __syncthreads();
    for (int i = 0; i < 4; ++i) {
        int n = n0 + ty + i * 8;
        dst[(size_t)n * K + (k0 + tx)] = (__bf16)tile[tx][ty + i * 8];
    }
}

// ------------- RMSNorm: fp32 in -> bf16 out -------------
__global__ __launch_bounds__(256) void rmsnorm_k(const float* __restrict__ x,
                                                 const float* __restrict__ w,
                                                 __bf16* __restrict__ out) {
    __shared__ float red[8];
    int row = blockIdx.x;
    const float* xr = x + (size_t)row * E_;
    float ss = 0.f;
    float vals[16];
    for (int j = 0; j < 16; ++j) {
        int c = threadIdx.x + j * 256;
        vals[j] = xr[c];
        ss += vals[j] * vals[j];
    }
    for (int off = 1; off < 32; off <<= 1) ss += __shfl_xor(ss, off, 32);
    int lane = threadIdx.x & 31, wv = threadIdx.x >> 5;
    if (lane == 0) red[wv] = ss;
    __syncthreads();
    float tot = 0.f;
    for (int i = 0; i < 8; ++i) tot += red[i];
    float rs = rsqrtf(tot / (float)E_ + EPSV);
    __bf16* orow = out + (size_t)row * E_;
    for (int j = 0; j < 16; ++j) {
        int c = threadIdx.x + j * 256;
        orow[c] = (__bf16)(vals[j] * rs * w[c]);
    }
}

// ------------- WMMA GEMM: C[M,N] = A[M,K] @ Bt[N,K]^T (+Res) -------------
// block = 256 threads = 8 waves; block tile 128x256, wave tile 64x64.
// A slab (128x32 bf16) staged in LDS via async Global->LDS copies (double
// buffered, ASYNCcnt); B fragments manually software-pipelined one k-step
// ahead in registers so the per-step barrier never stalls the WMMA stream:
// at compute time the B data was issued a full iteration earlier.
template <int ADD_RES>
__global__ __launch_bounds__(256) void gemm_bf16(const __bf16* __restrict__ A,
                                                 const __bf16* __restrict__ Bt,
                                                 const float* __restrict__ Res,
                                                 float* __restrict__ C,
                                                 int M, int N, int K) {
    __shared__ __align__(16) __bf16 ablk[2][128 * 32];
    int lane = threadIdx.x & 31;
    int wave = threadIdx.x >> 5;
    int rowband = (wave >> 2) * 64;
    int row0 = blockIdx.y * 128;
    int col0 = blockIdx.x * 256 + (wave & 3) * 64;
    int tid  = threadIdx.x;

    auto fill = [&](int buf, int k0) {
        // 128x32 bf16 = 512 x 16B chunks; 2 per thread
        for (int c = tid; c < 512; c += 256) {
            int rr = c >> 2, seg = c & 3;
            async_copy_b128(A + (size_t)(row0 + rr) * K + k0 + seg * 8,
                            &ablk[buf][rr * 32 + seg * 8]);
        }
    };

    const v8f zf = {0.f,0.f,0.f,0.f,0.f,0.f,0.f,0.f};
    v8f acc[4][4];
    for (int i = 0; i < 4; ++i) for (int j = 0; j < 4; ++j) acc[i][j] = zf;

    int r = lane & 15, half = lane >> 4;
    int nk = K / 32;

    fill(0, 0);                                   // async A slab for t=0
    Frag b[4], bn[4];
    for (int j = 0; j < 4; ++j) b[j] = load_b_frag(Bt, K, col0 + 16 * j, 0, lane);

    for (int t = 0; t < nk; ++t) {
        int k0 = t * 32, cur = t & 1;
        wait_async0();          // slab for this step landed
        __syncthreads();        // visible to all waves / prev slab fully consumed
        if (t + 1 < nk) {
            fill(1 - cur, k0 + 32);               // async A for t+1 (overlaps compute)
            for (int j = 0; j < 4; ++j)           // B for t+1, issued before WMMAs
                bn[j] = load_b_frag(Bt, K, col0 + 16 * j, k0 + 32, lane);
            __builtin_prefetch(Bt + (size_t)(col0 + r) * K + k0 + 64, 0, 0);
        }
        Frag a[4];
        for (int i = 0; i < 4; ++i) {             // ds_load_b128 from shared slab
            const __bf16* p = &ablk[cur][(rowband + 16 * i + r) * 32 + half * 8];
            a[i].h[0] = *(const v8bf*)p;
            a[i].h[1] = *(const v8bf*)(p + 16);
        }
        for (int i = 0; i < 4; ++i)
            for (int j = 0; j < 4; ++j)
                acc[i][j] = wmma_bf16(a[i], b[j], acc[i][j]);
        for (int j = 0; j < 4; ++j) b[j] = bn[j];
    }
    for (int i = 0; i < 4; ++i)
        for (int j = 0; j < 4; ++j)
            for (int e = 0; e < 8; ++e) {
                int row = row0 + rowband + i * 16 + e + 8 * half;
                int col = col0 + j * 16 + r;
                float v = acc[i][j][e];
                if (ADD_RES) v += Res[(size_t)row * N + col];
                C[(size_t)row * N + col] = v;
            }
}

// ------------- fused gate/up GEMM with SiLU epilogue, bf16 out -------------
__global__ __launch_bounds__(256) void gemm_gateup(const __bf16* __restrict__ A,
                                                   const __bf16* __restrict__ Bg,
                                                   const __bf16* __restrict__ Bu,
                                                   __bf16* __restrict__ Out,
                                                   int M, int N, int K) {
    int lane = threadIdx.x & 31;
    int wave = threadIdx.x >> 5;
    int row0 = blockIdx.y * 64 + (wave >> 2) * 32;
    int col0 = blockIdx.x * 256 + (wave & 3) * 64;
    const v8f zf = {0.f,0.f,0.f,0.f,0.f,0.f,0.f,0.f};
    v8f g[2][4], u[2][4];
    for (int i = 0; i < 2; ++i) for (int j = 0; j < 4; ++j) { g[i][j] = zf; u[i][j] = zf; }

    for (int k0 = 0; k0 < K; k0 += 32) {
        Frag a[2], bg[4], bu[4];
        for (int i = 0; i < 2; ++i) a[i]  = load_a_frag(A, K, row0 + 16 * i, k0, lane);
        for (int j = 0; j < 4; ++j) bg[j] = load_b_frag(Bg, K, col0 + 16 * j, k0, lane);
        for (int j = 0; j < 4; ++j) bu[j] = load_b_frag(Bu, K, col0 + 16 * j, k0, lane);
        for (int i = 0; i < 2; ++i)
            for (int j = 0; j < 4; ++j) {
                g[i][j] = wmma_bf16(a[i], bg[j], g[i][j]);
                u[i][j] = wmma_bf16(a[i], bu[j], u[i][j]);
            }
    }
    int r = lane & 15, half = lane >> 4;
    for (int i = 0; i < 2; ++i)
        for (int j = 0; j < 4; ++j)
            for (int e = 0; e < 8; ++e) {
                int row = row0 + i * 16 + e + 8 * half;
                int col = col0 + j * 16 + r;
                float gv = g[i][j][e], uv = u[i][j][e];
                float s  = gv / (1.f + __expf(-gv));    // SiLU (v_exp_f32)
                Out[(size_t)row * N + col] = (__bf16)(s * uv);
            }
}

// ------------- RoPE + pack Q,K to bf16 -------------
__global__ __launch_bounds__(256) void rope_qk(const float* __restrict__ qkv,
                                               const int* __restrict__ pos_ids,
                                               __bf16* __restrict__ Qb,
                                               __bf16* __restrict__ Kb) {
    int idx = blockIdx.x * blockDim.x + threadIdx.x;     // S * 40 heads * 64 pairs
    int s  = idx / ((NH_ + KVH_) * (HD_ / 2));
    int rm = idx % ((NH_ + KVH_) * (HD_ / 2));
    int hh = rm / (HD_ / 2);
    int i  = rm % (HD_ / 2);
    float pos = (float)pos_ids[s];
    float inv_freq = __powf(10000.0f, -(float)i / 64.0f);
    float ang = pos * inv_freq;
    float c = __cosf(ang), sn = __sinf(ang);
    const float* base; __bf16* out;
    if (hh < NH_) {
        base = qkv + (size_t)s * QKV_N + hh * HD_;
        out  = Qb  + (size_t)s * (NH_ * HD_) + hh * HD_;
    } else {
        int kh = hh - NH_;
        base = qkv + (size_t)s * QKV_N + NH_ * HD_ + kh * HD_;
        out  = Kb  + (size_t)s * (KVH_ * HD_) + kh * HD_;
    }
    float te = base[2 * i], to = base[2 * i + 1];
    out[2 * i]     = (__bf16)(te * c - to * sn);
    out[2 * i + 1] = (__bf16)(te * sn + to * c);
}

// V fp32 [s][kvh*128] -> Vt bf16 [kvh*128][s]  (B-source layout for PV WMMA)
__global__ __launch_bounds__(256) void v_transpose(const float* __restrict__ qkv,
                                                   __bf16* __restrict__ Vt) {
    int idx = blockIdx.x * blockDim.x + threadIdx.x;     // S * 1024
    int s = idx / (KVH_ * HD_);
    int d = idx % (KVH_ * HD_);
    Vt[(size_t)d * S_ + s] = (__bf16)qkv[(size_t)s * QKV_N + NH_ * HD_ + KVH_ * HD_ + d];
}

// ------------- flash attention (WMMA), one wave per (head, 16-row q tile) -----
__global__ __launch_bounds__(256) void attention_k(const __bf16* __restrict__ Qb,
                                                   const __bf16* __restrict__ Kb,
                                                   const __bf16* __restrict__ Vt,
                                                   __bf16* __restrict__ Ob) {
    __shared__ __align__(16) __bf16 plds[8][16][40];     // per-wave P tile, padded
    int lane = threadIdx.x & 31, wv = threadIdx.x >> 5;
    int gw   = blockIdx.x * 8 + wv;                      // 4096 waves total
    int head = gw >> 7;                                  // S/16 = 128 tiles/head
    int q0   = (gw & 127) * 16;
    int hkv  = head >> 2;                                // GQA group of 4
    const float scale = 0.08838834764831845f;            // 1/sqrt(128)
    int r = lane & 15, half = lane >> 4;

    Frag qf[4];
    for (int d = 0; d < 4; ++d)
        qf[d] = load_a_frag(Qb, NH_ * HD_, q0, head * HD_ + d * 32, lane);

    const v8f zf = {0.f,0.f,0.f,0.f,0.f,0.f,0.f,0.f};
    v8f o[8];
    float mrow[8], lrow[8];
    for (int e = 0; e < 8; ++e) { o[e] = zf; mrow[e] = -1e30f; lrow[e] = 0.f; }

    int kend = q0 + 16;
    for (int kc = 0; kc < kend; kc += 32) {
        // scores: two 16x16 tiles covering k columns [kc, kc+32)
        v8f s0 = zf, s1 = zf;
        for (int d = 0; d < 4; ++d) {
            Frag b0 = load_b_frag_cl(Kb, KVH_ * HD_, kc,      hkv * HD_ + d * 32, lane, S_ - 1);
            Frag b1 = load_b_frag_cl(Kb, KVH_ * HD_, kc + 16, hkv * HD_ + d * 32, lane, S_ - 1);
            s0 = wmma_bf16(qf[d], b0, s0);
            s1 = wmma_bf16(qf[d], b1, s1);
        }
        // scale + causal mask + online softmax (row = 16-lane half of wave)
        float p0[8], p1[8], cf[8];
        for (int e = 0; e < 8; ++e) {
            int rowg = q0 + e + 8 * half;
            float v0 = s0[e] * scale; if (kc + r       > rowg) v0 = -1e30f;
            float v1 = s1[e] * scale; if (kc + 16 + r  > rowg) v1 = -1e30f;
            float cm = fmaxf(v0, v1);
            for (int off = 1; off < 16; off <<= 1) cm = fmaxf(cm, __shfl_xor(cm, off, 32));
            float mn = fmaxf(mrow[e], cm);
            float cc = __expf(mrow[e] - mn);
            mrow[e] = mn;
            p0[e] = __expf(v0 - mn);
            p1[e] = __expf(v1 - mn);
            float rs = p0[e] + p1[e];
            for (int off = 1; off < 16; off <<= 1) rs += __shfl_xor(rs, off, 32);
            lrow[e] = lrow[e] * cc + rs;
            cf[e] = cc;
        }
        for (int ds = 0; ds < 8; ++ds)
            for (int e = 0; e < 8; ++e) o[ds][e] *= cf[e];
        // P (C-layout) -> LDS -> A-layout fragment
        for (int e = 0; e < 8; ++e) {
            int rl = e + 8 * half;
            plds[wv][rl][r]      = (__bf16)p0[e];
            plds[wv][rl][16 + r] = (__bf16)p1[e];
        }
        Frag pf;
        const __bf16* pp = &plds[wv][r][half * 16];
        pf.h[0] = *(const v8bf*)pp;
        pf.h[1] = *(const v8bf*)(pp + 8);
        // O += P @ V   (8 d-segments of 16)
        int cb = kc + half * 16; if (cb > S_ - 16) cb = S_ - 16;
        for (int ds = 0; ds < 8; ++ds) {
            const __bf16* vp = Vt + (size_t)(hkv * HD_ + ds * 16 + r) * S_ + cb;
            Frag vf;
            vf.h[0] = *(const v8bf*)vp;
            vf.h[1] = *(const v8bf*)(vp + 8);
            o[ds] = wmma_bf16(pf, vf, o[ds]);
        }
    }
    // normalize + store bf16 [S][H*D]
    for (int ds = 0; ds < 8; ++ds)
        for (int e = 0; e < 8; ++e) {
            int rowg = q0 + e + 8 * half;
            int col  = head * HD_ + ds * 16 + r;
            Ob[(size_t)rowg * (NH_ * HD_) + col] = (__bf16)(o[ds][e] / lrow[e]);
        }
}

// ---------------------------------------------------------------------------
extern "C" void kernel_launch(void* const* d_in, const int* in_sizes, int n_in,
                              void* d_out, int out_size, void* d_ws, size_t ws_size,
                              hipStream_t stream) {
    const float* x      = (const float*)d_in[0];
    const int*   pos    = (const int*)  d_in[1];
    const float* ln_w   = (const float*)d_in[2];
    const float* ffln_w = (const float*)d_in[3];
    const float* w_q    = (const float*)d_in[4];
    const float* w_k    = (const float*)d_in[5];
    const float* w_v    = (const float*)d_in[6];
    const float* w_d    = (const float*)d_in[7];
    const float* w_g    = (const float*)d_in[8];
    const float* w_u    = (const float*)d_in[9];
    const float* w_dn   = (const float*)d_in[10];
    float* out = (float*)d_out;

    // ---- workspace carve-up (256B aligned) ----
    uint8_t* ws = (uint8_t*)d_ws;
    size_t off = 0;
    auto carve = [&](size_t bytes) { uint8_t* p = ws + off; off = (off + bytes + 255) & ~(size_t)255; return p; };
    __bf16* wt_qkv  = (__bf16*)carve((size_t)QKV_N * E_   * 2);   // [6144][4096]
    __bf16* wt_den  = (__bf16*)carve((size_t)E_    * E_   * 2);   // [4096][4096]
    __bf16* wt_gate = (__bf16*)carve((size_t)HID_  * E_   * 2);   // [11008][4096]
    __bf16* wt_up   = (__bf16*)carve((size_t)HID_  * E_   * 2);
    __bf16* wt_down = (__bf16*)carve((size_t)E_    * HID_ * 2);   // [4096][11008]
    __bf16* h1      = (__bf16*)carve((size_t)S_ * E_    * 2);
    float*  qkv     = (float*) carve((size_t)S_ * QKV_N * 4);
    __bf16* Qb      = (__bf16*)carve((size_t)S_ * NH_  * HD_ * 2);
    __bf16* Kb      = (__bf16*)carve((size_t)S_ * KVH_ * HD_ * 2);
    __bf16* Vt      = (__bf16*)carve((size_t)KVH_ * HD_ * S_ * 2);
    __bf16* attnb   = (__bf16*)carve((size_t)S_ * E_   * 2);
    float*  x2      = (float*) carve((size_t)S_ * E_   * 4);
    __bf16* h2      = (__bf16*)carve((size_t)S_ * E_   * 2);
    __bf16* ffb     = (__bf16*)carve((size_t)S_ * HID_ * 2);

    dim3 tb(32, 8);
    // ---- weights: transpose + fp32->bf16 (one-time streaming pass) ----
    transpose_cvt<<<dim3(E_/32,   E_/32),   tb, 0, stream>>>(w_q,  wt_qkv,                         E_,   E_);
    transpose_cvt<<<dim3(KVH_*HD_/32, E_/32), tb, 0, stream>>>(w_k,  wt_qkv + (size_t)(NH_*HD_)*E_, E_,   KVH_*HD_);
    transpose_cvt<<<dim3(KVH_*HD_/32, E_/32), tb, 0, stream>>>(w_v,  wt_qkv + (size_t)(NH_*HD_+KVH_*HD_)*E_, E_, KVH_*HD_);
    transpose_cvt<<<dim3(E_/32,   E_/32),   tb, 0, stream>>>(w_d,  wt_den,  E_,   E_);
    transpose_cvt<<<dim3(HID_/32, E_/32),   tb, 0, stream>>>(w_g,  wt_gate, E_,   HID_);
    transpose_cvt<<<dim3(HID_/32, E_/32),   tb, 0, stream>>>(w_u,  wt_up,   E_,   HID_);
    transpose_cvt<<<dim3(E_/32,   HID_/32), tb, 0, stream>>>(w_dn, wt_down, HID_, E_);

    // ---- attention half ----
    rmsnorm_k<<<S_, 256, 0, stream>>>(x, ln_w, h1);
    gemm_bf16<0><<<dim3(QKV_N/256, S_/128), 256, 0, stream>>>(h1, wt_qkv, nullptr, qkv, S_, QKV_N, E_);
    {
        int tot = S_ * (NH_ + KVH_) * (HD_ / 2);
        rope_qk<<<tot / 256, 256, 0, stream>>>(qkv, pos, Qb, Kb);
        v_transpose<<<(S_ * KVH_ * HD_) / 256, 256, 0, stream>>>(qkv, Vt);
    }
    attention_k<<<(NH_ * (S_/16)) / 8, 256, 0, stream>>>(Qb, Kb, Vt, attnb);
    gemm_bf16<1><<<dim3(E_/256, S_/128), 256, 0, stream>>>(attnb, wt_den, x, x2, S_, E_, E_);

    // ---- MLP half ----
    rmsnorm_k<<<S_, 256, 0, stream>>>(x2, ffln_w, h2);
    gemm_gateup<<<dim3(HID_/256, S_/64), 256, 0, stream>>>(h2, wt_gate, wt_up, ffb, S_, HID_, E_);
    gemm_bf16<1><<<dim3(E_/256, S_/128), 256, 0, stream>>>(ffb, wt_down, x2, out, S_, E_, HID_);

    (void)in_sizes; (void)n_in; (void)out_size; (void)ws_size;
}
